// V2TemporalPoseTransformer_63797444215498
// MI455X (gfx1250) — matrix operand order repaired and checked by reading, not compiled, not run on hardware
//
#include <hip/hip_runtime.h>
#include <hip/hip_bf16.h>
#include <math.h>

#define BB   2
#define TT   2048
#define DD   512
#define HH   8
#define LL   4
#define DFF  2048
#define WW   256
#define DHD  64
#define MR   (BB * TT)      // 4096 rows of activations
#define EPSF 1e-5f

typedef __attribute__((ext_vector_type(16))) __bf16 v16bf;
typedef __attribute__((ext_vector_type(8)))  __bf16 v8bf;
typedef __attribute__((ext_vector_type(8)))  float  v8f;

// ---------------------------------------------------------------- cast fp32 -> bf16 (activations)
__global__ void cast_f32_to_bf16(const float* __restrict__ src, __bf16* __restrict__ dst, int n) {
    int i = blockIdx.x * blockDim.x + threadIdx.x;
    if (i < n) dst[i] = (__bf16)src[i];
}

// ---------------------------------------------------------------- pack weights fp32 -> bf16 WMMA-B fragments
// Source: L matrices of (K,N) row-major fp32.
// Dest layout: [l][kt][nt][lane][e] bf16, where the fragment for k-tile kt / n-tile nt
// is 32 lanes x 16 elements contiguous; element e of lane maps to
//   K = kt*32 + 8*(lane>>4) + (e&7) + 16*(e>>3),  N = nt*16 + (lane&15)      (ISA 16-bit B 32x16 layout)
// GEMM then loads each B fragment as one 32-byte v16bf per lane (coalesced b128).
__global__ void pack_weights_wmma(const float* __restrict__ src, __bf16* __restrict__ dst,
                                  int K, int N, int total) {
    int i = blockIdx.x * blockDim.x + threadIdx.x;
    if (i >= total) return;
    int KN = K * N;
    int l  = i / KN;
    int r  = i % KN;
    int e    = r & 15;
    int lane = (r >> 4) & 31;
    int rest = r >> 9;                   // / (16*32)
    int ntiles = N >> 4;
    int nt = rest % ntiles;
    int kt = rest / ntiles;
    int krow = kt * 32 + ((lane >> 4) << 3) + (e & 7) + ((e >> 3) << 4);
    int ncol = nt * 16 + (lane & 15);
    dst[i] = (__bf16)src[(size_t)l * KN + (size_t)krow * N + ncol];
}

// ---------------------------------------------------------------- layernorm (row per block, 256 thr, D=512)
__global__ void layernorm_kernel(const float* __restrict__ x,
                                 const float* __restrict__ g,
                                 const float* __restrict__ b,
                                 __bf16* __restrict__ outb,
                                 float*  __restrict__ outf) {
    int row = blockIdx.x;
    int tid = threadIdx.x;                       // 0..255
    const float* xr = x + (size_t)row * DD;
    float v0 = xr[tid], v1 = xr[tid + 256];
    __shared__ float sh[256], sh2[256];
    sh[tid]  = v0 + v1;
    sh2[tid] = v0 * v0 + v1 * v1;
    __syncthreads();
    for (int off = 128; off > 0; off >>= 1) {
        if (tid < off) { sh[tid] += sh[tid + off]; sh2[tid] += sh2[tid + off]; }
        __syncthreads();
    }
    float mean = sh[0] * (1.0f / DD);
    float var  = sh2[0] * (1.0f / DD) - mean * mean;
    float rstd = rsqrtf(var + EPSF);
    float y0 = (v0 - mean) * rstd * g[tid]       + b[tid];
    float y1 = (v1 - mean) * rstd * g[tid + 256] + b[tid + 256];
    size_t base = (size_t)row * DD;
    if (outb) {
        outb[base + tid]       = (__bf16)y0;
        outb[base + tid + 256] = (__bf16)y1;
    } else {
        outf[base + tid]       = y0;
        outf[base + tid + 256] = y1;
    }
}

// ---------------------------------------------------------------- WMMA GEMM
// Y(M,N) = A(M,K)[bf16 row-major] * B(K,N)[bf16, pre-packed fragments] + bias
// MODE 0: outf = Y          (fp32)
// MODE 1: outf += Y         (fused residual add)
// MODE 2: outb = gelu(Y)    (bf16, exact erf GELU)
// grid = (M/32, N/64), block = 32 (one wave); wave computes a 32x64 tile:
// 2 A fragments x 4 B fragments -> 8 WMMAs per 32-wide K step.
template <int MODE>
__global__ void gemm_wmma_kernel(const __bf16* __restrict__ A,
                                 const __bf16* __restrict__ Bp,
                                 const float*  __restrict__ bias,
                                 float* __restrict__ outf,
                                 __bf16* __restrict__ outb,
                                 int K, int N) {
    int m0   = blockIdx.x * 32;
    int n0   = blockIdx.y * 64;
    int lane = threadIdx.x;                      // 0..31
    int ncol = lane & 15;
    int hi   = lane >> 4;                        // 0 or 1
    int ntiles = N >> 4;
    int nt0  = n0 >> 4;

    v8f acc[2][4];
    #pragma unroll
    for (int mi = 0; mi < 2; ++mi)
        #pragma unroll
        for (int t = 0; t < 4; ++t)
            #pragma unroll
            for (int r = 0; r < 8; ++r) acc[mi][t][r] = 0.0f;

    const __bf16* arow0 = A + (size_t)(m0 + ncol) * K;        // rows m0+ncol, m0+16+ncol
    const __bf16* arow1 = arow0 + (size_t)16 * K;

    for (int k0 = 0; k0 < K; k0 += 32) {
        // ---- A fragments: 16-bit A 16x32 layout (ISA 7.12.2)
        // lanes 0-15: K = {k0+0..7, k0+16..23}; lanes 16-31: K = {k0+8..15, k0+24..31}
        v16bf a[2];
        #pragma unroll
        for (int mi = 0; mi < 2; ++mi) {
            const __bf16* ap = (mi == 0 ? arow0 : arow1) + k0 + (hi << 3);
            v8bf alo = *(const v8bf*)ap;
            v8bf ahi = *(const v8bf*)(ap + 16);
            #pragma unroll
            for (int e = 0; e < 8; ++e) { a[mi][e] = alo[e]; a[mi][e + 8] = ahi[e]; }
        }
        // ---- B fragments: one contiguous 32B vector per lane per fragment
        const __bf16* bbase = Bp + ((((size_t)(k0 >> 5) * ntiles + nt0) * 32) + lane) * 16;
        v16bf bm[4];
        #pragma unroll
        for (int t = 0; t < 4; ++t)
            bm[t] = *(const v16bf*)(bbase + (size_t)t * 32 * 16);

        #pragma unroll
        for (int mi = 0; mi < 2; ++mi)
            #pragma unroll
            for (int t = 0; t < 4; ++t)
                acc[mi][t] = __builtin_amdgcn_wmma_f32_16x16x32_bf16(
                    false, a[mi], false, bm[t], (short)0, acc[mi][t], false, false);
    }

    // ---- store: VGPR r -> M = 8*hi + r, N = ncol (ISA C/D layout)
    #pragma unroll
    for (int mi = 0; mi < 2; ++mi) {
        int mbase = m0 + mi * 16 + (hi << 3);
        #pragma unroll
        for (int t = 0; t < 4; ++t) {
            int n = n0 + t * 16 + ncol;
            float bv = bias[n];
            #pragma unroll
            for (int r = 0; r < 8; ++r) {
                float val = acc[mi][t][r] + bv;
                size_t idx = (size_t)(mbase + r) * N + n;
                if (MODE == 0) {
                    outf[idx] = val;
                } else if (MODE == 1) {
                    outf[idx] += val;
                } else {
                    float ge = 0.5f * val * (1.0f + erff(val * 0.70710678118654752f));
                    outb[idx] = (__bf16)ge;
                }
            }
        }
    }
}

// ---------------------------------------------------------------- RoPE (in place on q and k, fp32 (B*T, D))
__global__ void rope_kernel(float* __restrict__ q, float* __restrict__ k) {
    const int half = DHD / 2;                    // 32
    int i = blockIdx.x * blockDim.x + threadIdx.x;
    if (i >= MR * HH * half) return;
    int j   = i % half;
    int rem = i / half;
    int h   = rem % HH;
    int bt  = rem / HH;
    int t   = bt % TT;
    size_t base = (size_t)bt * DD + h * DHD;
    float inv = powf(10000.0f, -(float)(2 * j) / (float)DHD);
    float ang = (float)t * inv;
    float c = cosf(ang), s = sinf(ang);
    float q0 = q[base + j], q1 = q[base + j + half];
    q[base + j]        = q0 * c - q1 * s;
    q[base + j + half] = q1 * c + q0 * s;
    float k0 = k[base + j], k1 = k[base + j + half];
    k[base + j]        = k0 * c - k1 * s;
    k[base + j + half] = k1 * c + k0 * s;
}

// ---------------------------------------------------------------- sliding-window causal attention
// One block (256 threads) per query (b,h,t); window length <= W = 256 -> one key per thread.
__global__ void attn_kernel(const float* __restrict__ q,
                            const float* __restrict__ k,
                            const float* __restrict__ v,
                            __bf16* __restrict__ out) {
    int gid = blockIdx.x;                        // (b*H + h)*T + t
    int t   = gid % TT;
    int bh  = gid / TT;
    int h   = bh % HH;
    int b   = bh / HH;
    int tid = threadIdx.x;                       // 0..255

    __shared__ float qs[DHD];
    __shared__ float ps[256];
    __shared__ float red[256];

    size_t qoff = ((size_t)(b * TT + t)) * DD + h * DHD;
    if (tid < DHD) qs[tid] = q[qoff + tid];
    __syncthreads();

    int jstart = t - (WW - 1); if (jstart < 0) jstart = 0;
    int jpos   = jstart + tid;

    float sc = -INFINITY;
    if (jpos <= t) {
        const float* kr = k + ((size_t)(b * TT + jpos)) * DD + h * DHD;
        float acc = 0.0f;
        #pragma unroll 8
        for (int d = 0; d < DHD; ++d) acc += qs[d] * kr[d];
        sc = acc * 0.125f;                       // 1/sqrt(64)
    }

    red[tid] = sc; __syncthreads();
    for (int off = 128; off > 0; off >>= 1) {
        if (tid < off) red[tid] = fmaxf(red[tid], red[tid + off]);
        __syncthreads();
    }
    float mx = red[0];
    __syncthreads();

    float p = (sc == -INFINITY) ? 0.0f : expf(sc - mx);
    ps[tid]  = p;
    red[tid] = p; __syncthreads();
    for (int off = 128; off > 0; off >>= 1) {
        if (tid < off) red[tid] += red[tid + off];
        __syncthreads();
    }
    float invs = 1.0f / red[0];

    if (tid < DHD) {
        float acc = 0.0f;
        int jcount = t - jstart + 1;
        for (int jj = 0; jj < jcount; ++jj) {
            acc += ps[jj] * v[((size_t)(b * TT + jstart + jj)) * DD + h * DHD + tid];
        }
        out[qoff + tid] = (__bf16)(acc * invs);
    }
}

// ---------------------------------------------------------------- launch
extern "C" void kernel_launch(void* const* d_in, const int* in_sizes, int n_in,
                              void* d_out, int out_size, void* d_ws, size_t ws_size,
                              hipStream_t stream) {
    const float* tokens = (const float*)d_in[0];
    const float* Wq  = (const float*)d_in[1];
    const float* Wk  = (const float*)d_in[2];
    const float* Wv  = (const float*)d_in[3];
    const float* Wo  = (const float*)d_in[4];
    const float* bq  = (const float*)d_in[5];
    const float* bk  = (const float*)d_in[6];
    const float* bv  = (const float*)d_in[7];
    const float* bo  = (const float*)d_in[8];
    const float* W1  = (const float*)d_in[9];
    const float* b1  = (const float*)d_in[10];
    const float* W2  = (const float*)d_in[11];
    const float* b2  = (const float*)d_in[12];
    const float* g1  = (const float*)d_in[13];
    const float* be1 = (const float*)d_in[14];
    const float* g2  = (const float*)d_in[15];
    const float* be2 = (const float*)d_in[16];
    const float* gf  = (const float*)d_in[17];
    const float* bf  = (const float*)d_in[18];
    float* outp = (float*)d_out;

    // ---- workspace bump allocator
    char* p = (char*)d_ws;
    auto alloc = [&](size_t bytes) -> char* {
        char* r = p;
        p += (bytes + 255) & ~(size_t)255;
        return r;
    };
    float*  x     = (float*) alloc((size_t)MR * DD * 4);
    __bf16* xn_b  = (__bf16*)alloc((size_t)MR * DD * 2);
    float*  qb    = (float*) alloc((size_t)MR * DD * 4);
    float*  kb    = (float*) alloc((size_t)MR * DD * 4);
    float*  vb    = (float*) alloc((size_t)MR * DD * 4);
    __bf16* at_b  = (__bf16*)alloc((size_t)MR * DD * 2);
    __bf16* h_b   = (__bf16*)alloc((size_t)MR * DFF * 2);
    __bf16* wq_b  = (__bf16*)alloc((size_t)LL * DD * DD * 2);
    __bf16* wk_b  = (__bf16*)alloc((size_t)LL * DD * DD * 2);
    __bf16* wv_b  = (__bf16*)alloc((size_t)LL * DD * DD * 2);
    __bf16* wo_b  = (__bf16*)alloc((size_t)LL * DD * DD * 2);
    __bf16* w1_b  = (__bf16*)alloc((size_t)LL * DD * DFF * 2);
    __bf16* w2_b  = (__bf16*)alloc((size_t)LL * DFF * DD * 2);

    // ---- weight packs fp32 -> bf16 WMMA fragments
    {
        int nDD = LL * DD * DD;
        int nDF = LL * DD * DFF;
        int blk = 256;
        pack_weights_wmma<<<(nDD + blk - 1) / blk, blk, 0, stream>>>(Wq, wq_b, DD, DD, nDD);
        pack_weights_wmma<<<(nDD + blk - 1) / blk, blk, 0, stream>>>(Wk, wk_b, DD, DD, nDD);
        pack_weights_wmma<<<(nDD + blk - 1) / blk, blk, 0, stream>>>(Wv, wv_b, DD, DD, nDD);
        pack_weights_wmma<<<(nDD + blk - 1) / blk, blk, 0, stream>>>(Wo, wo_b, DD, DD, nDD);
        pack_weights_wmma<<<(nDF + blk - 1) / blk, blk, 0, stream>>>(W1, w1_b, DD, DFF, nDF);
        pack_weights_wmma<<<(nDF + blk - 1) / blk, blk, 0, stream>>>(W2, w2_b, DFF, DD, nDF);
    }

    // ---- x = tokens
    hipMemcpyAsync(x, tokens, (size_t)MR * DD * 4, hipMemcpyDeviceToDevice, stream);

    dim3 gemmBlk(32, 1, 1);
    dim3 gD(MR / 32, DD / 64, 1);    // N = 512
    dim3 gF(MR / 32, DFF / 64, 1);   // N = 2048
    int ropeN = MR * HH * (DHD / 2);
    dim3 ropeG((ropeN + 255) / 256, 1, 1);
    dim3 attnG(BB * HH * TT, 1, 1);

    for (int l = 0; l < LL; ++l) {
        size_t wDD = (size_t)l * DD * DD;
        size_t wDF = (size_t)l * DD * DFF;
        // LN1 -> bf16
        layernorm_kernel<<<MR, 256, 0, stream>>>(x, g1 + l * DD, be1 + l * DD, xn_b, nullptr);
        // Q, K, V projections
        gemm_wmma_kernel<0><<<gD, gemmBlk, 0, stream>>>(xn_b, wq_b + wDD, bq + l * DD, qb, nullptr, DD, DD);
        gemm_wmma_kernel<0><<<gD, gemmBlk, 0, stream>>>(xn_b, wk_b + wDD, bk + l * DD, kb, nullptr, DD, DD);
        gemm_wmma_kernel<0><<<gD, gemmBlk, 0, stream>>>(xn_b, wv_b + wDD, bv + l * DD, vb, nullptr, DD, DD);
        // RoPE on q, k
        rope_kernel<<<ropeG, 256, 0, stream>>>(qb, kb);
        // windowed attention -> bf16
        attn_kernel<<<attnG, 256, 0, stream>>>(qb, kb, vb, at_b);
        // O projection, fused residual add
        gemm_wmma_kernel<1><<<gD, gemmBlk, 0, stream>>>(at_b, wo_b + wDD, bo + l * DD, x, nullptr, DD, DD);
        // LN2 -> bf16
        layernorm_kernel<<<MR, 256, 0, stream>>>(x, g2 + l * DD, be2 + l * DD, xn_b, nullptr);
        // FFN up + exact GELU -> bf16
        gemm_wmma_kernel<2><<<gF, gemmBlk, 0, stream>>>(xn_b, w1_b + wDF, b1 + l * DFF, nullptr, h_b, DD, DFF);
        // FFN down, fused residual add
        gemm_wmma_kernel<1><<<gD, gemmBlk, 0, stream>>>(h_b, w2_b + wDF, b2 + l * DD, x, nullptr, DFF, DD);
    }

    // ---- final layernorm -> fp32 output
    layernorm_kernel<<<MR, 256, 0, stream>>>(x, gf, bf, nullptr, outp);
}
